// ContrastiveLoss_6674379178082
// MI455X (gfx1250) — matrix operand order repaired and checked by reading, not compiled, not run on hardware
//
#include <hip/hip_runtime.h>
#include <hip/hip_bf16.h>

typedef __attribute__((ext_vector_type(16))) _Float16 v16h;
typedef __attribute__((ext_vector_type(8)))  _Float16 v8h;
typedef __attribute__((ext_vector_type(4)))  _Float16 v4h;
typedef __attribute__((ext_vector_type(8)))  float    v8f;
typedef __attribute__((ext_vector_type(4)))  float    v4f;

#define TWO_N   8192      // 2 * BATCH_SIZE
#define HALF_N  4096      // BATCH_SIZE
#define DIM     512       // PROJ_DIM
#define INV_T   2.0f      // 1 / TEMPERATURE
#define EPS     1e-8f

#define BSTRIDE 520       // LDS row stride in halves (512 + 8 pad -> bank-safe)
#define NTILES  (TWO_N / 16)

// ---------------------------------------------------------------------------
// Kernel 1: L2-normalize rows of [z_i; z_j] and convert to fp16.
// One wave (32 lanes) per row; each lane owns 16 floats (4 x float4).
// ---------------------------------------------------------------------------
__global__ void nxent_normalize(const float* __restrict__ zi,
                                const float* __restrict__ zj,
                                _Float16* __restrict__ reps) {
  const int lane = threadIdx.x & 31;
  const int wave = threadIdx.x >> 5;
  const int row  = blockIdx.x * 8 + wave;           // 8 waves / block

  const float* src = (row < HALF_N) ? (zi + (size_t)row * DIM)
                                    : (zj + (size_t)(row - HALF_N) * DIM);
  v4f vals[4];
  float ss = 0.0f;
#pragma unroll
  for (int c = 0; c < 4; ++c) {
    vals[c] = *(const v4f*)(src + c * 128 + lane * 4);
    ss += vals[c].x * vals[c].x + vals[c].y * vals[c].y +
          vals[c].z * vals[c].z + vals[c].w * vals[c].w;
  }
#pragma unroll
  for (int m = 16; m >= 1; m >>= 1) ss += __shfl_xor(ss, m, 32);

  const float rn = 1.0f / fmaxf(sqrtf(ss), EPS);

  _Float16* dst = reps + (size_t)row * DIM;
#pragma unroll
  for (int c = 0; c < 4; ++c) {
    v4h h;
    h.x = (_Float16)(vals[c].x * rn);
    h.y = (_Float16)(vals[c].y * rn);
    h.z = (_Float16)(vals[c].z * rn);
    h.w = (_Float16)(vals[c].w * rn);
    *(v4h*)(dst + c * 128 + lane * 4) = h;
  }
}

// ---------------------------------------------------------------------------
// Kernel 2: fused sim = R*R^T (WMMA f16 -> f32) + exp-sum + positive pick.
// 4 waves/block, each wave owns a 16-row stripe (A kept entirely in VGPRs,
// launch_bounds gives the allocator room so nothing spills). The B tile for
// each column step is staged cooperatively into double-buffered LDS and
// shared by all 4 waves; the next tile's global fetch overlaps compute.
// ---------------------------------------------------------------------------
__global__ void __launch_bounds__(128, 1)
nxent_simloss(const _Float16* __restrict__ reps,
              float* __restrict__ rowloss) {
  __shared__ _Float16 bt[2][16 * BSTRIDE];          // 2 x 16.25 KB

  const int tid  = threadIdx.x;
  const int lane = tid & 31;
  const int wave = tid >> 5;
  const int i0   = (blockIdx.x * 4 + wave) * 16;    // grid 128, 64 rows/block
  const int m    = lane & 15;                       // M (for A) / N (for B)
  const int hi   = lane >> 4;                       // half-wave selector

  // ---- A fragments: rows i0..i0+15, all K=512 in VGPRs (16 x v16h).
  // A 16x32 layout (contiguous K, matching B): lane<16 holds K {0..15}; lane>=16 holds
  // K {16..31}  => two contiguous 16B loads per 32-K chunk.
  v16h afrag[16];
  {
    const _Float16* arow = reps + (size_t)(i0 + m) * DIM;
#pragma unroll
    for (int kc = 0; kc < 16; ++kc) {
      union { v16h v; v8h h[2]; } u;
      u.h[0] = *(const v8h*)(arow + kc * 32 + hi * 16);
      u.h[1] = *(const v8h*)(arow + kc * 32 + hi * 16 + 8);
      afrag[kc] = u.v;
    }
  }

  // ---- cooperative B-tile staging geometry: 128 threads cover 16 rows x
  // 512 halves; thread -> (row r, 64-half segment seg), 4 x 32B copies.
  const int srow = tid >> 3;                        // 0..15
  const int sseg = tid & 7;                         // 0..7

  float acc[8];   // per-row running sum of exp(logits) (self excluded)
  float pos[8];   // per-row positive logit
#pragma unroll
  for (int r = 0; r < 8; ++r) { acc[r] = 0.0f; pos[r] = 0.0f; }

  const int rowbase = i0 + hi * 8;   // f32 C/D layout: VGPR r = row rowbase+r

  // ---- preload tile 0 into buffer 0
  v16h fr[4];
  {
    const _Float16* g = reps + (size_t)srow * DIM + sseg * 64;
#pragma unroll
    for (int c = 0; c < 4; ++c) fr[c] = *(const v16h*)(g + c * 16);
    _Float16* l = &bt[0][srow * BSTRIDE + sseg * 64];
#pragma unroll
    for (int c = 0; c < 4; ++c) *(v16h*)(l + c * 16) = fr[c];
  }
  __syncthreads();

  for (int t = 0; t < NTILES; ++t) {
    const int j0 = t * 16;

    // ---- kick off the next tile's global fetch (overlaps WMMA below)
    if (t + 1 < NTILES) {
      const _Float16* g = reps + (size_t)(j0 + 16 + srow) * DIM + sseg * 64;
      __builtin_prefetch(g + 16 * DIM, 0, 0);       // tile after next -> L2
#pragma unroll
      for (int c = 0; c < 4; ++c) fr[c] = *(const v16h*)(g + c * 16);
    }

    // ---- 16x16 output tile: 16 chained WMMAs over K=512, B from LDS.
    // B fragment (32xK x 16xN): lane n reads 16 contiguous halves at
    // kc*32 + hi*16 of LDS row n  => two ds_load_b128, bank-conflict-free.
    const _Float16* brow = &bt[t & 1][m * BSTRIDE + hi * 16];
    v8f c = {};
#pragma unroll
    for (int kc = 0; kc < 16; ++kc) {
      v16h b = *(const v16h*)(brow + kc * 32);
      c = __builtin_amdgcn_wmma_f32_16x16x32_f16(
          /*neg_a=*/false, afrag[kc], /*neg_b=*/false, b,
          /*c_mod=*/(short)0, c, /*reuse_a=*/false, /*reuse_b=*/false);
    }

    // ---- consume the tile: exp-sum (self excluded) + positive logit
    const int col = j0 + m;
#pragma unroll
    for (int r = 0; r < 8; ++r) {
      const int row  = rowbase + r;
      const float lg = c[r] * INV_T;
      float e = __expf(lg);
      if (col == row) e = 0.0f;                 // exclude self-similarity
      acc[r] += e;
      if (col == (row ^ HALF_N)) pos[r] += lg;  // positive: (row+N) % 2N
    }

    // ---- rotate double buffer
    __syncthreads();
    if (t + 1 < NTILES) {
      _Float16* l = &bt[(t + 1) & 1][srow * BSTRIDE + sseg * 64];
#pragma unroll
      for (int cc = 0; cc < 4; ++cc) *(v16h*)(l + cc * 16) = fr[cc];
    }
    __syncthreads();
  }

  // ---- reduce across the 16-lane halves; lanes m==0 hold the row totals.
#pragma unroll
  for (int r = 0; r < 8; ++r) {
    float a = acc[r], p = pos[r];
#pragma unroll
    for (int msk = 8; msk >= 1; msk >>= 1) {
      a += __shfl_xor(a, msk, 32);
      p += __shfl_xor(p, msk, 32);
    }
    if (m == 0) {
      // loss_partial = -(pos/T - log(denom)) = log(denom) - pos/T
      rowloss[rowbase + r] = __logf(a) - p;
    }
  }
}

// ---------------------------------------------------------------------------
// Kernel 3: mean over the 8192 per-row losses.
// ---------------------------------------------------------------------------
__global__ void nxent_reduce(const float* __restrict__ rowloss,
                             float* __restrict__ out) {
  __shared__ float sm[256];
  float s = 0.0f;
  for (int i = threadIdx.x; i < TWO_N; i += 256) s += rowloss[i];
  sm[threadIdx.x] = s;
  __syncthreads();
#pragma unroll
  for (int off = 128; off > 0; off >>= 1) {
    if ((int)threadIdx.x < off) sm[threadIdx.x] += sm[threadIdx.x + off];
    __syncthreads();
  }
  if (threadIdx.x == 0) out[0] = sm[0] / (float)TWO_N;
}

// ---------------------------------------------------------------------------
extern "C" void kernel_launch(void* const* d_in, const int* in_sizes, int n_in,
                              void* d_out, int out_size, void* d_ws, size_t ws_size,
                              hipStream_t stream) {
  const float* zi = (const float*)d_in[0];   // [4096, 512] f32
  const float* zj = (const float*)d_in[1];   // [4096, 512] f32
  float* out = (float*)d_out;                // scalar loss

  // Workspace layout: fp16 reps (8 MB) | per-row loss (32 KB)
  _Float16* reps    = (_Float16*)d_ws;
  float*    rowloss = (float*)((char*)d_ws + (size_t)TWO_N * DIM * sizeof(_Float16));

  // 1) normalize + fp16 convert: 8192 rows, 1 wave/row, 8 waves/block
  nxent_normalize<<<TWO_N / 8, 256, 0, stream>>>(zi, zj, reps);

  // 2) fused sim/softmax-denominator: 128 blocks x 4 waves (64 rows/block)
  nxent_simloss<<<TWO_N / 64, 128, 0, stream>>>(reps, rowloss);

  // 3) final mean
  nxent_reduce<<<1, 256, 0, stream>>>(rowloss, out);
}
